// CausalSelfAttention_35012573397207
// MI455X (gfx1250) — matrix operand order repaired and checked by reading, not compiled, not run on hardware
//
#include <hip/hip_runtime.h>
#include <hip/hip_bf16.h>

// ---------------------------------------------------------------------------
// Causal self-attention for MI455X (gfx1250), wave32, WMMA bf16 pipeline.
//   B=4, T=2048, C=1024, H=16, hs=64.  M = B*T = 8192.
// Kernel 1: qkv = x @ W_attn + b_attn   (WMMA bf16, scatter to [B,H,T,hs] bf16)
// Kernel 2: flash attention (async-to-LDS K tiles, WMMA QK^T + online softmax
//           + WMMA P*V)
// Kernel 3: out = y @ W_proj + b_proj   (WMMA bf16, f32 out)
//
// GEMMs: 128x128 block tile, 8 waves in a 2x4 grid, 64x32 per wave =
// 8 WMMAs from 6 fragments per K-step.  All LDS tiles are staged in
// "fragment-ready" layouts so each WMMA operand is two contiguous 16B chunks
// per lane (ds_load_b128, no scalar u16 gathers).
// ---------------------------------------------------------------------------

typedef __attribute__((ext_vector_type(16))) __bf16 v16bf;
typedef __attribute__((ext_vector_type(8)))  __bf16 v8bf;
typedef __attribute__((ext_vector_type(2)))  __bf16 v2bf;
typedef __attribute__((ext_vector_type(8)))  float  v8f;

struct bfpair { uint4 a, b; };   // 32B == v16bf

// Native f32 -> bf16 (fptrunc, RNE): backend picks the hw cvt op.
__device__ __forceinline__ __bf16 f2bf(float f) { return (__bf16)f; }

// Pack two f32 -> two bf16 in one dword (packed cvt when available).
__device__ __forceinline__ unsigned f2bf2(float lo, float hi) {
#if __has_builtin(__builtin_amdgcn_cvt_pk_bf16_f32)
  auto p = __builtin_amdgcn_cvt_pk_bf16_f32(lo, hi);
  return __builtin_bit_cast(unsigned, p);
#else
  v2bf p;
  p[0] = (__bf16)lo;
  p[1] = (__bf16)hi;
  return __builtin_bit_cast(unsigned, p);
#endif
}

// A fragment (16x32) in this code's contiguous k order: lanes 0-15 hold K 0..15,
// lanes 16-31 hold K 16..31: two contiguous 8-element chunks per lane.
__device__ __forceinline__ v16bf ld_fragA(const __bf16* rowp, int hi) {
  bfpair f;
  f.a = *(const uint4*)(rowp + hi * 16);
  f.b = *(const uint4*)(rowp + hi * 16 + 8);
  return __builtin_bit_cast(v16bf, f);
}

// B fragment, same contiguous k order as A: lanes 0-15 hold K=0..15, lanes 16-31 hold K=16..31,
// contiguous when the tile is stored K-innermost: p = &tile[n][hi*16].
__device__ __forceinline__ v16bf ld_fragB(const __bf16* p) {
  bfpair f;
  f.a = *(const uint4*)(p);
  f.b = *(const uint4*)(p + 8);
  return __builtin_bit_cast(v16bf, f);
}

#define WMMA_BF16(A, B, C) \
  __builtin_amdgcn_wmma_f32_16x16x32_bf16(false, (A), false, (B), (short)0, (C), false, false)

// ---------------------------------------------------------------------------
// Kernel 1: QKV projection.  X[8192,1024] f32, W[1024,3072] f32 -> q/k/v bf16
// in [B,H,T,hs] layout.  Block 128x128, 8 waves of 64x32 (4x2 WMMA tiles).
// ---------------------------------------------------------------------------
__global__ __launch_bounds__(256) void qkv_gemm_kernel(
    const float* __restrict__ X, const float* __restrict__ W,
    const float* __restrict__ bias,
    __bf16* __restrict__ Qo, __bf16* __restrict__ Ko, __bf16* __restrict__ Vo) {
  __shared__ __bf16 Al[128][32];    // row-major (A frags are row-contiguous)
  __shared__ __bf16 Blt[128][32];   // TRANSPOSED: Blt[n][k] (B frags contiguous)

  const int tid  = threadIdx.x;
  const int lane = tid & 31;
  const int wave = tid >> 5;     // 0..7
  const int wm   = wave & 1;     // M direction (2)
  const int wn   = wave >> 1;    // N direction (4)
  const int hi   = lane >> 4;
  const int ln   = lane & 15;
  const int bm   = blockIdx.y * 128;
  const int bn   = blockIdx.x * 128;

  v8f acc[4][2] = {};

  for (int k0 = 0; k0 < 1024; k0 += 32) {
    {  // stage A tile 128x32 (f32 -> packed bf16, 2x b128 store per thread)
      int row = tid >> 1;
      int cb  = (tid & 1) * 16;
      const float4* src = (const float4*)(X + (size_t)(bm + row) * 1024 + k0 + cb);
      float4 f0 = src[0], f1 = src[1], f2 = src[2], f3 = src[3];
      uint4 pa, pb;
      pa.x = f2bf2(f0.x, f0.y); pa.y = f2bf2(f0.z, f0.w);
      pa.z = f2bf2(f1.x, f1.y); pa.w = f2bf2(f1.z, f1.w);
      pb.x = f2bf2(f2.x, f2.y); pb.y = f2bf2(f2.z, f2.w);
      pb.z = f2bf2(f3.x, f3.y); pb.w = f2bf2(f3.z, f3.w);
      *(uint4*)&Al[row][cb]     = pa;
      *(uint4*)&Al[row][cb + 8] = pb;
    }
    {  // stage B tile 32x128 transposed into Blt[n][k], packed k-pairs
      int k2 = (tid >> 4) * 2;     // even k row: 0..30
      int cb = (tid & 15) * 8;     // n chunk: 0..120
      const float4* r0 = (const float4*)(W + (size_t)(k0 + k2) * 3072 + bn + cb);
      const float4* r1 = (const float4*)(W + (size_t)(k0 + k2 + 1) * 3072 + bn + cb);
      float4 a0 = r0[0], a1 = r0[1];
      float4 b0 = r1[0], b1 = r1[1];
      *(unsigned*)&Blt[cb + 0][k2] = f2bf2(a0.x, b0.x);
      *(unsigned*)&Blt[cb + 1][k2] = f2bf2(a0.y, b0.y);
      *(unsigned*)&Blt[cb + 2][k2] = f2bf2(a0.z, b0.z);
      *(unsigned*)&Blt[cb + 3][k2] = f2bf2(a0.w, b0.w);
      *(unsigned*)&Blt[cb + 4][k2] = f2bf2(a1.x, b1.x);
      *(unsigned*)&Blt[cb + 5][k2] = f2bf2(a1.y, b1.y);
      *(unsigned*)&Blt[cb + 6][k2] = f2bf2(a1.z, b1.z);
      *(unsigned*)&Blt[cb + 7][k2] = f2bf2(a1.w, b1.w);
    }
    __syncthreads();

    v16bf af[4], bfr[2];
#pragma unroll
    for (int i = 0; i < 4; ++i)
      af[i] = ld_fragA(&Al[wm * 64 + i * 16 + ln][0], hi);
#pragma unroll
    for (int j = 0; j < 2; ++j)
      bfr[j] = ld_fragB(&Blt[wn * 32 + j * 16 + ln][hi * 16]);
#pragma unroll
    for (int i = 0; i < 4; ++i)
#pragma unroll
      for (int j = 0; j < 2; ++j)
        acc[i][j] = WMMA_BF16(af[i], bfr[j], acc[i][j]);
    __syncthreads();
  }

  // Epilogue: bias add, scatter bf16 to q/k/v in [B,H,T,hs]
#pragma unroll
  for (int j = 0; j < 2; ++j) {
    int ncol  = bn + wn * 32 + j * 16 + ln;   // 0..3071
    int which = ncol >> 10;
    int c     = ncol & 1023;
    int h     = c >> 6, d = c & 63;
    __bf16* dst = (which == 0) ? Qo : (which == 1) ? Ko : Vo;
    float bv = bias[ncol];
#pragma unroll
    for (int i = 0; i < 4; ++i) {
#pragma unroll
      for (int r = 0; r < 8; ++r) {
        int m  = bm + wm * 64 + i * 16 + r + 8 * hi;
        int bb = m >> 11, t = m & 2047;
        dst[(((size_t)bb * 16 + h) * 2048 + t) * 64 + d] = f2bf(acc[i][j][r] + bv);
      }
    }
  }
}

// ---------------------------------------------------------------------------
// Kernel 2: flash attention.  One block per (b, h, 128-query tile); each wave
// owns 16 query rows.  K tiles DMA'd via global_load_async_to_lds_b128;
// V tiles staged transposed; causal tile skip.
// ---------------------------------------------------------------------------
__global__ __launch_bounds__(256) void flash_attn_kernel(
    const __bf16* __restrict__ Qg, const __bf16* __restrict__ Kg,
    const __bf16* __restrict__ Vg, float* __restrict__ Yg) {
  __shared__ __bf16 Kl[64][64];      // row-major [k][d]  (K frags row-contig)
  __shared__ __bf16 Vlt[64][64];     // TRANSPOSED [d][k] (V frags contiguous)
  __shared__ __bf16 Pl[8][16][64];   // per-wave P tile, row-major [q][k]

  const int tid  = threadIdx.x;
  const int lane = tid & 31;
  const int wave = tid >> 5;
  const int hi   = lane >> 4;
  const int ln   = lane & 15;

  const int bid = blockIdx.x;          // B*H*(T/128) = 1024
  const int qt  = bid & 15;
  const int h   = (bid >> 4) & 15;
  const int b   = bid >> 8;
  const size_t bh = ((size_t)b * 16 + h) * 2048;   // row base into [BH,T,64]
  const int q0 = qt * 128 + wave * 16;

  // Q fragments resident in VGPRs: 16x64 bf16 per wave, A-layout (4x b128)
  v16bf qf[2];
  {
    const __bf16* qrow = Qg + (bh + q0 + ln) * 64;
    qf[0] = ld_fragA(qrow, hi);
    qf[1] = ld_fragA(qrow + 32, hi);
  }

  v8f o[4] = {};
  float mrow[8], lrow[8];
#pragma unroll
  for (int r = 0; r < 8; ++r) { mrow[r] = -1e30f; lrow[r] = 0.0f; }

  const int kr  = tid >> 2;          // staging row 0..63
  const int seg = (tid & 3) * 16;    // staging col chunk

  const int nkt = qt * 2 + 2;  // causal: only tiles with keys <= block max query
  for (int kt = 0; kt < nkt; ++kt) {
    {  // --- stage K tile via CDNA5 async DMA to LDS (32B per thread) -------
      unsigned lds0 = (unsigned)(uintptr_t)&Kl[kr][seg];
      unsigned goff = (unsigned)(((bh + (size_t)kt * 64 + kr) * 64 + seg) *
                                 sizeof(__bf16));
      asm volatile("global_load_async_to_lds_b128 %0, %1, %2"
                   :: "v"(lds0), "v"(goff), "s"((const void*)Kg) : "memory");
      asm volatile("global_load_async_to_lds_b128 %0, %1, %2"
                   :: "v"(lds0 + 16u), "v"(goff + 16u), "s"((const void*)Kg)
                   : "memory");
      // --- stage V tile transposed into Vlt[d][k] ------------------------
      const uint4* vs = (const uint4*)(Vg + (bh + (size_t)kt * 64 + kr) * 64 + seg);
      v8bf va = __builtin_bit_cast(v8bf, vs[0]);
      v8bf vb = __builtin_bit_cast(v8bf, vs[1]);
#pragma unroll
      for (int i = 0; i < 8; ++i) {
        Vlt[seg + i][kr]     = va[i];
        Vlt[seg + 8 + i][kr] = vb[i];
      }
      if (kt + 1 < nkt) {  // prefetch next K/V tile into GL2
        __builtin_prefetch(Kg + (bh + (size_t)(kt + 1) * 64 + kr) * 64 + seg, 0, 1);
        __builtin_prefetch(Vg + (bh + (size_t)(kt + 1) * 64 + kr) * 64 + seg, 0, 1);
      }
      asm volatile("s_wait_asynccnt 0x0" ::: "memory");  // async K in LDS
    }
    __syncthreads();

    if (kt * 64 <= q0 + 15) {  // wave-level causal skip
      // S = Q * K^T : 4 tiles of (16q x 16k), inner dim 64 = 2 WMMAs each
      v8f s[4];
#pragma unroll
      for (int t = 0; t < 4; ++t) {
        const __bf16* krow = &Kl[t * 16 + ln][0];  // B[d,n] = K[n,d], d-contig
        v16bf kf0 = ld_fragB(krow + hi * 16);
        v16bf kf1 = ld_fragB(krow + 32 + hi * 16);
        v8f cc = {};
        cc = WMMA_BF16(qf[0], kf0, cc);
        cc = WMMA_BF16(qf[1], kf1, cc);
        s[t] = cc;
      }

      // scale, causal mask, online softmax (rows span 16-lane half-waves)
#pragma unroll
      for (int r = 0; r < 8; ++r) {
        const int qglob = q0 + r + 8 * hi;
        float mx = -1e30f;
#pragma unroll
        for (int t = 0; t < 4; ++t) {
          int kg = kt * 64 + t * 16 + ln;
          float vsc = s[t][r] * 0.125f;                 // 1/sqrt(64)
          vsc = (kg <= qglob) ? vsc : -1e30f;
          s[t][r] = vsc;
          mx = fmaxf(mx, vsc);
        }
#pragma unroll
        for (int msk = 1; msk < 16; msk <<= 1) mx = fmaxf(mx, __shfl_xor(mx, msk, 32));
        float mnew  = fmaxf(mrow[r], mx);
        float alpha = __expf(mrow[r] - mnew);
        float sum = 0.0f;
#pragma unroll
        for (int t = 0; t < 4; ++t) {
          float p = __expf(s[t][r] - mnew);
          s[t][r] = p;
          sum += p;
        }
#pragma unroll
        for (int msk = 1; msk < 16; msk <<= 1) sum += __shfl_xor(sum, msk, 32);
        lrow[r] = lrow[r] * alpha + sum;
        mrow[r] = mnew;
#pragma unroll
        for (int dt = 0; dt < 4; ++dt) o[dt][r] *= alpha;
      }

      // P (C-layout f32) -> LDS (bf16) -> A-layout fragments
#pragma unroll
      for (int r = 0; r < 8; ++r)
#pragma unroll
        for (int t = 0; t < 4; ++t)
          Pl[wave][r + 8 * hi][t * 16 + ln] = f2bf(s[t][r]);
      asm volatile("s_wait_dscnt 0x0" ::: "memory");    // CDNA5 split DS counter

      const __bf16* prow = &Pl[wave][ln][0];
      v16bf pf0 = ld_fragA(prow, hi);
      v16bf pf1 = ld_fragA(prow + 32, hi);

      // O += P * V : 4 d-tiles, inner dim 64 = 2 WMMAs each (Vlt contiguous)
#pragma unroll
      for (int dt = 0; dt < 4; ++dt) {
        const __bf16* vrow = &Vlt[dt * 16 + ln][0];
        v16bf vf0 = ld_fragB(vrow + hi * 16);
        v16bf vf1 = ld_fragB(vrow + 32 + hi * 16);
        o[dt] = WMMA_BF16(pf0, vf0, o[dt]);
        o[dt] = WMMA_BF16(pf1, vf1, o[dt]);
      }
    }
    __syncthreads();
  }

  // normalize and write y back in [B,T,C] (f32) for the output projection
#pragma unroll
  for (int r = 0; r < 8; ++r) {
    const int qglob = q0 + r + 8 * hi;
    const float inv = 1.0f / lrow[r];
#pragma unroll
    for (int dt = 0; dt < 4; ++dt)
      Yg[((size_t)b * 2048 + qglob) * 1024 + h * 64 + dt * 16 + ln] = o[dt][r] * inv;
  }
}

// ---------------------------------------------------------------------------
// Kernel 3: output projection.  Y[8192,1024] f32 x W_proj[1024,1024] + b.
// Block 128x128, 8 waves of 64x32.
// ---------------------------------------------------------------------------
__global__ __launch_bounds__(256) void proj_gemm_kernel(
    const float* __restrict__ Y, const float* __restrict__ W,
    const float* __restrict__ bias, float* __restrict__ Out) {
  __shared__ __bf16 Al[128][32];
  __shared__ __bf16 Blt[128][32];   // transposed, fragment-ready

  const int tid  = threadIdx.x;
  const int lane = tid & 31;
  const int wave = tid >> 5;
  const int wm   = wave & 1;
  const int wn   = wave >> 1;
  const int hi   = lane >> 4;
  const int ln   = lane & 15;
  const int bm   = blockIdx.y * 128;
  const int bn   = blockIdx.x * 128;

  v8f acc[4][2] = {};

  for (int k0 = 0; k0 < 1024; k0 += 32) {
    {
      int row = tid >> 1;
      int cb  = (tid & 1) * 16;
      const float4* src = (const float4*)(Y + (size_t)(bm + row) * 1024 + k0 + cb);
      float4 f0 = src[0], f1 = src[1], f2 = src[2], f3 = src[3];
      uint4 pa, pb;
      pa.x = f2bf2(f0.x, f0.y); pa.y = f2bf2(f0.z, f0.w);
      pa.z = f2bf2(f1.x, f1.y); pa.w = f2bf2(f1.z, f1.w);
      pb.x = f2bf2(f2.x, f2.y); pb.y = f2bf2(f2.z, f2.w);
      pb.z = f2bf2(f3.x, f3.y); pb.w = f2bf2(f3.z, f3.w);
      *(uint4*)&Al[row][cb]     = pa;
      *(uint4*)&Al[row][cb + 8] = pb;
    }
    {
      int k2 = (tid >> 4) * 2;
      int cb = (tid & 15) * 8;
      const float4* r0 = (const float4*)(W + (size_t)(k0 + k2) * 1024 + bn + cb);
      const float4* r1 = (const float4*)(W + (size_t)(k0 + k2 + 1) * 1024 + bn + cb);
      float4 a0 = r0[0], a1 = r0[1];
      float4 b0 = r1[0], b1 = r1[1];
      *(unsigned*)&Blt[cb + 0][k2] = f2bf2(a0.x, b0.x);
      *(unsigned*)&Blt[cb + 1][k2] = f2bf2(a0.y, b0.y);
      *(unsigned*)&Blt[cb + 2][k2] = f2bf2(a0.z, b0.z);
      *(unsigned*)&Blt[cb + 3][k2] = f2bf2(a0.w, b0.w);
      *(unsigned*)&Blt[cb + 4][k2] = f2bf2(a1.x, b1.x);
      *(unsigned*)&Blt[cb + 5][k2] = f2bf2(a1.y, b1.y);
      *(unsigned*)&Blt[cb + 6][k2] = f2bf2(a1.z, b1.z);
      *(unsigned*)&Blt[cb + 7][k2] = f2bf2(a1.w, b1.w);
    }
    __syncthreads();

    v16bf af[4], bfr[2];
#pragma unroll
    for (int i = 0; i < 4; ++i)
      af[i] = ld_fragA(&Al[wm * 64 + i * 16 + ln][0], hi);
#pragma unroll
    for (int j = 0; j < 2; ++j)
      bfr[j] = ld_fragB(&Blt[wn * 32 + j * 16 + ln][hi * 16]);
#pragma unroll
    for (int i = 0; i < 4; ++i)
#pragma unroll
      for (int j = 0; j < 2; ++j)
        acc[i][j] = WMMA_BF16(af[i], bfr[j], acc[i][j]);
    __syncthreads();
  }

#pragma unroll
  for (int j = 0; j < 2; ++j) {
    int ncol = bn + wn * 32 + j * 16 + ln;
    float bv = bias[ncol];
#pragma unroll
    for (int i = 0; i < 4; ++i) {
#pragma unroll
      for (int r = 0; r < 8; ++r) {
        int m = bm + wm * 64 + i * 16 + r + 8 * hi;
        Out[(size_t)m * 1024 + ncol] = acc[i][j][r] + bv;
      }
    }
  }
}

// ---------------------------------------------------------------------------
extern "C" void kernel_launch(void* const* d_in, const int* in_sizes, int n_in,
                              void* d_out, int out_size, void* d_ws, size_t ws_size,
                              hipStream_t stream) {
  const float* x      = (const float*)d_in[0];
  const float* W_attn = (const float*)d_in[1];
  const float* b_attn = (const float*)d_in[2];
  const float* W_proj = (const float*)d_in[3];
  const float* b_proj = (const float*)d_in[4];
  float* out = (float*)d_out;

  // Workspace layout (bytes):
  //   q bf16 [B,H,T,64] : 16 MiB @ 0
  //   k bf16            : 16 MiB @ 16 MiB
  //   v bf16            : 16 MiB @ 32 MiB
  //   y f32  [B,T,C]    : 32 MiB @ 48 MiB     (total 80 MiB)
  char* ws = (char*)d_ws;
  __bf16* Qb = (__bf16*)(ws);
  __bf16* Kb = (__bf16*)(ws + (size_t)16 * 1024 * 1024);
  __bf16* Vb = (__bf16*)(ws + (size_t)32 * 1024 * 1024);
  float*  Yb = (float*)(ws + (size_t)48 * 1024 * 1024);

  // 1) QKV projection: M=8192, N=3072, block tiles 128x128
  qkv_gemm_kernel<<<dim3(3072 / 128, 8192 / 128), 256, 0, stream>>>(
      x, W_attn, b_attn, Qb, Kb, Vb);

  // 2) Flash attention: B*H*(T/128) blocks
  flash_attn_kernel<<<dim3(4 * 16 * (2048 / 128)), 256, 0, stream>>>(Qb, Kb, Vb, Yb);

  // 3) Output projection: M=8192, N=1024, block tiles 128x128
  proj_gemm_kernel<<<dim3(1024 / 128, 8192 / 128), 256, 0, stream>>>(
      Yb, W_proj, b_proj, out);
}
